// MoCo_75247827026350
// MI455X (gfx1250) — matrix operand order repaired and checked by reading, not compiled, not run on hardware
//
#include <hip/hip_runtime.h>

typedef __attribute__((ext_vector_type(16))) __bf16 v16bf;
typedef __attribute__((ext_vector_type(8)))  float  v8f;

#define B_ROWS 4096
#define D_IN   1024
#define DD     256
#define KK     1024
#define LQ     32768
#define QK_ROWS 4096
#define TEMP_INV (1.0f/0.07f)
#define GTEMP_INV 2.0f
#define LOSS_WAVES 4
#define TILE_BYTES 8192   // 16 rows x 256 bf16

// ---------------- WMMA fragment helpers (bf16, 16x16x32) ----------------

// A-matrix fragment (16 rows x 32 K), from f32 source, convert on the fly.
__device__ inline v16bf load_A_frag_f32(const float* base, int ld, int row0, int k0) {
  const int lane = threadIdx.x & 31;
  const int m = lane & 15;
  const int koff = (lane < 16) ? 0 : 8;
  const float* p = base + (size_t)(row0 + m) * ld + k0;
  v16bf a;
#pragma unroll
  for (int j = 0; j < 8; ++j) a[j]     = (__bf16)p[koff + j];
#pragma unroll
  for (int j = 0; j < 8; ++j) a[8 + j] = (__bf16)p[16 + koff + j];
  return a;
}

__device__ inline v16bf load_A_frag_bf(const __bf16* base, int ld, int row0, int k0) {
  const int lane = threadIdx.x & 31;
  const int m = lane & 15;
  const int koff = (lane < 16) ? 0 : 8;
  const __bf16* p = base + (size_t)(row0 + m) * ld + k0;
  v16bf a;
#pragma unroll
  for (int j = 0; j < 8; ++j) a[j]     = p[koff + j];
#pragma unroll
  for (int j = 0; j < 8; ++j) a[8 + j] = p[16 + koff + j];
  return a;
}

// B-matrix fragment (32 K x 16 N) where B = X^T, X row-major [N, ld] bf16.
// lane n holds column n of B == row n of X; K order k = 8h+j, 16+8h+j (two runs of 8).
__device__ inline v16bf load_Bt_frag_bf(const __bf16* X, int ld, int n0, int k0) {
  const int lane = threadIdx.x & 31;
  const int n = lane & 15;
  const int kh = (lane < 16) ? 0 : 8;
  const __bf16* p = X + (size_t)(n0 + n) * ld + k0 + kh;
  v16bf b;
#pragma unroll
  for (int j = 0; j < 8; ++j) b[j] = p[j];
#pragma unroll
  for (int j = 0; j < 8; ++j) b[8 + j] = p[16 + j];
  return b;
}

__device__ inline void store_C(float* Out, int ld, int row0, int col0, v8f c) {
  const int lane = threadIdx.x & 31;
  const int n = lane & 15;
  const int rb = (lane < 16) ? 0 : 8;
#pragma unroll
  for (int i = 0; i < 8; ++i)
    Out[(size_t)(row0 + rb + i) * ld + col0 + n] = c[i];
}

#define WMMA_BF16(a, b, c) \
  __builtin_amdgcn_wmma_f32_16x16x32_bf16(false, (a), false, (b), (short)0, (c), false, false)

// async copy of one 16x256 bf16 tile (8KB) from global to LDS, 16 x b128/lane
__device__ inline void async_tile_16x256(const __bf16* gsrc, unsigned lds_base) {
  const int lane = threadIdx.x & 31;
  const unsigned long long gaddr = (unsigned long long)(uintptr_t)gsrc;
#pragma unroll
  for (int j = 0; j < 16; ++j) {
    const unsigned off = (unsigned)((j * 32 + lane) * 16);
    const unsigned dst = lds_base + off;
    asm volatile("global_load_async_to_lds_b128 %0, %1, %2"
                 :: "v"(dst), "v"(off), "s"(gaddr) : "memory");
  }
}

// ---------------- block reduction helpers ----------------

__device__ inline float block_reduce_sum256(float v) {
  __shared__ float red_s[256];
  red_s[threadIdx.x] = v; __syncthreads();
  for (int s = 128; s > 0; s >>= 1) {
    if ((int)threadIdx.x < s) red_s[threadIdx.x] += red_s[threadIdx.x + s];
    __syncthreads();
  }
  float r = red_s[0]; __syncthreads();
  return r;
}

__device__ inline float block_reduce_max256(float v) {
  __shared__ float red_m[256];
  red_m[threadIdx.x] = v; __syncthreads();
  for (int s = 128; s > 0; s >>= 1) {
    if ((int)threadIdx.x < s) red_m[threadIdx.x] = fmaxf(red_m[threadIdx.x], red_m[threadIdx.x + s]);
    __syncthreads();
  }
  float r = red_m[0]; __syncthreads();
  return r;
}

__device__ inline float gumbel_from_u(float u) {
  u = u * (1.0f - 2e-6f) + 1e-6f;
  return -__logf(-__logf(u));
}

// ---------------- normalization / staging kernels ----------------

__global__ void rownorm_bf16_kernel(const float* __restrict__ in, __bf16* __restrict__ out) {
  const int r = blockIdx.x;
  float x = in[(size_t)r * DD + threadIdx.x];
  float s = block_reduce_sum256(x * x);
  float sc = rsqrtf(fmaxf(s, 1e-12f));
  out[(size_t)r * DD + threadIdx.x] = (__bf16)(x * sc);
}

// queue = l2norm(2*l2norm(queue_k) + noise)  -> bf16
__global__ void queue_mix_norm_kernel(const float* __restrict__ qk, const float* __restrict__ noise,
                                      __bf16* __restrict__ out) {
  const int r = blockIdx.x;
  float x = qk[(size_t)r * DD + threadIdx.x];
  float s1 = block_reduce_sum256(x * x);
  float y = 2.0f * x * rsqrtf(fmaxf(s1, 1e-12f)) + noise[(size_t)r * DD + threadIdx.x];
  float s2 = block_reduce_sum256(y * y);
  float sc = rsqrtf(fmaxf(s2, 1e-12f));
  out[(size_t)r * DD + threadIdx.x] = (__bf16)(y * sc);
}

// Wt[n][k] = bf16(W[k][n]);  W is [1024,256] f32, Wt is [256,1024] bf16
__global__ void transpose_w_kernel(const float* __restrict__ W, __bf16* __restrict__ Wt) {
  const int k = blockIdx.x;
  const int n = threadIdx.x;
  Wt[(size_t)n * D_IN + k] = (__bf16)W[(size_t)k * DD + n];
}

// feat row-norm producing f32 + bf16 + transposed bf16 (+ optional d_out copy)
__global__ void feat_norm_kernel(const float* __restrict__ raw, float* __restrict__ outf,
                                 __bf16* __restrict__ outbf, __bf16* __restrict__ outT,
                                 float* __restrict__ out2) {
  const int r = blockIdx.x;
  float x = raw[(size_t)r * DD + threadIdx.x];
  float s = block_reduce_sum256(x * x);
  float y = x * rsqrtf(fmaxf(s, 1e-12f));
  outf[(size_t)r * DD + threadIdx.x] = y;
  outbf[(size_t)r * DD + threadIdx.x] = (__bf16)y;
  outT[(size_t)threadIdx.x * B_ROWS + r] = (__bf16)y;
  if (out2) out2[(size_t)r * DD + threadIdx.x] = y;
}

__global__ void agg_norm_kernel(const float* __restrict__ raw, const float* __restrict__ nrm,
                                float* __restrict__ outf, __bf16* __restrict__ outbf,
                                float* __restrict__ out2) {
  const int r = blockIdx.x;
  float x = raw[(size_t)r * DD + threadIdx.x] / nrm[r];
  float s = block_reduce_sum256(x * x);
  float y = x * rsqrtf(fmaxf(s, 1e-12f));
  outf[(size_t)r * DD + threadIdx.x] = y;
  outbf[(size_t)r * DD + threadIdx.x] = (__bf16)y;
  if (out2) out2[(size_t)r * DD + threadIdx.x] = y;
}

// ---------------- WMMA GEMM kernels (2 N-tiles per wave, A reused) ----------------

// feat_raw[4096,256] = x[4096,1024] @ W[1024,256]; B from Wt bf16 [256,1024]
__global__ void gemm_feat_kernel(const float* __restrict__ X, const __bf16* __restrict__ Wt,
                                 float* __restrict__ Out) {
  const int wave = threadIdx.x >> 5;
  const int pid = blockIdx.x * 8 + wave;           // 2048 wave-jobs = 256 x 8
  const int row0 = (pid >> 3) * 16;
  const int col0 = (pid & 7) * 32;
  v8f c0 = {}, c1 = {};
#pragma unroll 4
  for (int k0 = 0; k0 < D_IN; k0 += 32) {
    v16bf a  = load_A_frag_f32(X, D_IN, row0, k0);
    v16bf b0 = load_Bt_frag_bf(Wt, D_IN, col0, k0);
    v16bf b1 = load_Bt_frag_bf(Wt, D_IN, col0 + 16, k0);
    c0 = WMMA_BF16(a, b0, c0);
    c1 = WMMA_BF16(a, b1, c1);
  }
  store_C(Out, DD, row0, col0, c0);
  store_C(Out, DD, row0, col0 + 16, c1);
}

// logits[4096,1024] = feat_bf[4096,256] @ ctx_bf[1024,256]^T
__global__ void gemm_logits_kernel(const __bf16* __restrict__ feat, const __bf16* __restrict__ ctx,
                                   float* __restrict__ logits) {
  const int wave = threadIdx.x >> 5;
  const int pid = blockIdx.x * 8 + wave;           // 8192 wave-jobs = 256 x 32
  const int row0 = (pid >> 5) * 16;
  const int col0 = (pid & 31) * 32;
  v8f c0 = {}, c1 = {};
#pragma unroll
  for (int k0 = 0; k0 < DD; k0 += 32) {
    v16bf a  = load_A_frag_bf(feat, DD, row0, k0);
    v16bf b0 = load_Bt_frag_bf(ctx, DD, col0, k0);
    v16bf b1 = load_Bt_frag_bf(ctx, DD, col0 + 16, k0);
    c0 = WMMA_BF16(a, b0, c0);
    c1 = WMMA_BF16(a, b1, c1);
  }
  store_C(logits, KK, row0, col0, c0);
  store_C(logits, KK, row0, col0 + 16, c1);
}

// agg_raw[1024,256] = assignT_bf[1024,4096] @ feat[4096,256]; B from featT [256,4096]
__global__ void gemm_agg_kernel(const __bf16* __restrict__ assignT, const __bf16* __restrict__ featT,
                                float* __restrict__ Out) {
  const int wave = threadIdx.x >> 5;
  const int pid = blockIdx.x * 8 + wave;           // 512 wave-jobs = 64 x 8
  const int row0 = (pid >> 3) * 16;
  const int col0 = (pid & 7) * 32;
  v8f c0 = {}, c1 = {};
  for (int k0 = 0; k0 < B_ROWS; k0 += 32) {
    v16bf a  = load_A_frag_bf(assignT, B_ROWS, row0, k0);
    v16bf b0 = load_Bt_frag_bf(featT, B_ROWS, col0, k0);
    v16bf b1 = load_Bt_frag_bf(featT, B_ROWS, col0 + 16, k0);
    c0 = WMMA_BF16(a, b0, c0);
    c1 = WMMA_BF16(a, b1, c1);
  }
  store_C(Out, DD, row0, col0, c0);
  store_C(Out, DD, row0, col0 + 16, c1);
}

// ---------------- gumbel softmax / assignment ----------------

__global__ void gumbel_assign_kernel(const float* __restrict__ logits, const float* __restrict__ ua,
                                     const float* __restrict__ ub, __bf16* __restrict__ assignT,
                                     float* __restrict__ out_assign) {
  const int b = blockIdx.x;
  const int t = threadIdx.x;
  const float* lrow = logits + (size_t)b * KK;
  const float* uar  = ua + (size_t)b * KK;
  const float* ubr  = ub + (size_t)b * KK;
  float ys[4];
  float lmax = -1e30f, hval = -1e30f;
  int hidx = 0;
#pragma unroll
  for (int j = 0; j < 4; ++j) {
    const int k = t + j * 256;
    const float lg = lrow[k];
    const float y = (lg + gumbel_from_u(uar[k])) * GTEMP_INV;
    ys[j] = y; lmax = fmaxf(lmax, y);
    const float h = lg + gumbel_from_u(ubr[k]);   // argmax invariant to /temp
    if (h > hval) { hval = h; hidx = k; }
  }
  const float m = block_reduce_max256(lmax);
  float se = 0.0f;
#pragma unroll
  for (int j = 0; j < 4; ++j) { ys[j] = __expf(ys[j] - m); se += ys[j]; }
  const float s = block_reduce_sum256(se);
  const float inv = 1.0f / s;
#pragma unroll
  for (int j = 0; j < 4; ++j) {
    const int k = t + j * 256;
    assignT[(size_t)k * B_ROWS + b] = (__bf16)(ys[j] * inv);
  }
  __shared__ float hv[256]; __shared__ int hi[256];
  hv[t] = hval; hi[t] = hidx; __syncthreads();
  for (int s2 = 128; s2 > 0; s2 >>= 1) {
    if (t < s2) {
      if (hv[t + s2] > hv[t] || (hv[t + s2] == hv[t] && hi[t + s2] < hi[t])) {
        hv[t] = hv[t + s2]; hi[t] = hi[t + s2];
      }
    }
    __syncthreads();
  }
  if (t == 0 && out_assign) out_assign[b] = (float)hi[0];
}

__global__ void normalizer_kernel(const __bf16* __restrict__ assignT, float* __restrict__ nrm) {
  const int k = blockIdx.x;
  const __bf16* p = assignT + (size_t)k * B_ROWS;
  float s = 0.0f;
  for (int b = threadIdx.x; b < B_ROWS; b += 256) s += (float)p[b];
  s = block_reduce_sum256(s);
  if (threadIdx.x == 0) nrm[k] = s + 1e-8f;
}

// ---------------- fused WMMA online-softmax InfoNCE ----------------
// grid.x = Nrows/16, block = 128 (4 waves). Each wave streams queue tiles
// through a double-buffered async LDS pipeline and keeps per-row (max,sumexp).
__global__ void loss_ce_kernel(const float* __restrict__ qf, const __bf16* __restrict__ qbf,
                               const float* __restrict__ kf, const __bf16* __restrict__ queue_bf,
                               int L, float invN, float* __restrict__ loss_out) {
  __shared__ __align__(16) char smem[LOSS_WAVES * 2 * TILE_BYTES];   // 64 KB staging
  const int lane = threadIdx.x & 31;
  const int wave = threadIdx.x >> 5;
  const int row0 = blockIdx.x * 16;
  const unsigned sbase = (unsigned)(uintptr_t)(&smem[0]) + (unsigned)(wave * 2 * TILE_BYTES);

  v16bf afrag[8];
#pragma unroll
  for (int kk = 0; kk < 8; ++kk) afrag[kk] = load_A_frag_bf(qbf, DD, row0, kk * 32);

  float rm[8], rs[8];
#pragma unroll
  for (int i = 0; i < 8; ++i) { rm[i] = -1e30f; rs[i] = 0.0f; }

  const int ntiles = L >> 4;
  int st = 0;
  async_tile_16x256(queue_bf + (size_t)wave * 16 * DD, sbase);
  for (int t = wave; t < ntiles; t += LOSS_WAVES, st ^= 1) {
    const int tn = t + LOSS_WAVES;
    if (tn < ntiles) {
      async_tile_16x256(queue_bf + (size_t)tn * 16 * DD, sbase + (unsigned)((st ^ 1) * TILE_BYTES));
      asm volatile("s_wait_asynccnt 0x10" ::: "memory");   // oldest 16 (current tile) resident
    } else {
      asm volatile("s_wait_asynccnt 0x0" ::: "memory");
    }
    const __bf16* buf = (const __bf16*)(smem + wave * 2 * TILE_BYTES + st * TILE_BYTES);
    v8f c = {};
#pragma unroll
    for (int kk = 0; kk < 8; ++kk)
      c = WMMA_BF16(afrag[kk], load_Bt_frag_bf(buf, DD, 0, kk * 32), c);
#pragma unroll
    for (int i = 0; i < 8; ++i) {
      const float v = c[i] * TEMP_INV;
      if (v > rm[i]) { rs[i] = rs[i] * __expf(rm[i] - v) + 1.0f; rm[i] = v; }
      else           { rs[i] += __expf(v - rm[i]); }
    }
  }
  // reduce across the 16 lanes that share the same M-halves
#pragma unroll
  for (int msk = 8; msk >= 1; msk >>= 1) {
#pragma unroll
    for (int i = 0; i < 8; ++i) {
      const float om = __shfl_xor(rm[i], msk, 32);
      const float os = __shfl_xor(rs[i], msk, 32);
      const float nm = fmaxf(rm[i], om);
      rs[i] = rs[i] * __expf(rm[i] - nm) + os * __expf(om - nm);
      rm[i] = nm;
    }
  }
  __syncthreads();                     // all waves done reading stage buffers
  float* lm = (float*)smem;            // reuse staging LDS for the reduction
  float* ls = lm + LOSS_WAVES * 16;
  if (lane == 0) {
#pragma unroll
    for (int i = 0; i < 8; ++i) { lm[wave * 16 + i] = rm[i]; ls[wave * 16 + i] = rs[i]; }
  }
  if (lane == 16) {
#pragma unroll
    for (int i = 0; i < 8; ++i) { lm[wave * 16 + 8 + i] = rm[i]; ls[wave * 16 + 8 + i] = rs[i]; }
  }
  __syncthreads();
  if (threadIdx.x < 16) {
    const int r = threadIdx.x;
    float m = -1e30f;
#pragma unroll
    for (int w = 0; w < LOSS_WAVES; ++w) m = fmaxf(m, lm[w * 16 + r]);
    float s = 0.0f;
#pragma unroll
    for (int w = 0; w < LOSS_WAVES; ++w) s += ls[w * 16 + r] * __expf(lm[w * 16 + r] - m);
    const float* qr = qf + (size_t)(row0 + r) * DD;
    const float* kr = kf + (size_t)(row0 + r) * DD;
    float p = 0.0f;
    for (int j = 0; j < DD; ++j) p += qr[j] * kr[j];
    p *= TEMP_INV;
    const float M = fmaxf(m, p);
    const float S = s * __expf(m - M) + __expf(p - M);
    const float lse = M + __logf(S);
    atomicAdd(loss_out, (lse - p) * invN);
  }
}

__global__ void zero_kernel(float* p) { if (threadIdx.x == 0) p[0] = 0.0f; }

// ---------------- host launch ----------------

extern "C" void kernel_launch(void* const* d_in, const int* in_sizes, int n_in,
                              void* d_out, int out_size, void* d_ws, size_t ws_size,
                              hipStream_t stream) {
  const float* x1   = (const float*)d_in[0];
  const float* x2   = (const float*)d_in[1];
  const float* W1   = (const float*)d_in[2];
  const float* W2   = (const float*)d_in[3];
  const float* ctx1 = (const float*)d_in[4];
  const float* ctx2 = (const float*)d_in[5];
  const float* qn   = (const float*)d_in[6];
  const float* qk   = (const float*)d_in[7];
  const float* u1a  = (const float*)d_in[8];
  const float* u1b  = (const float*)d_in[9];
  const float* u2a  = (const float*)d_in[10];
  const float* u2b  = (const float*)d_in[11];
  const float* qnz  = (const float*)d_in[12];

  float* out = (float*)d_out;
  float* out_assign = out;                                   // [4096]
  float* out_aggn2  = out + B_ROWS;                          // [4096,256]
  float* out_aggk2  = out_aggn2 + (size_t)B_ROWS * DD;       // [1024,256]
  float* out_loss   = out_aggk2 + (size_t)KK * DD;           // [1]

  char* w = (char*)d_ws;
  auto alloc = [&](size_t bytes) -> char* {
    char* p = w; w += (bytes + 255) & ~(size_t)255; return p;
  };
  __bf16* ctx1_bf   = (__bf16*)alloc((size_t)KK * DD * 2);
  __bf16* ctx2_bf   = (__bf16*)alloc((size_t)KK * DD * 2);
  __bf16* qn_bf     = (__bf16*)alloc((size_t)LQ * DD * 2);
  __bf16* queue_bf  = (__bf16*)alloc((size_t)QK_ROWS * DD * 2);
  __bf16* Wt1_bf    = (__bf16*)alloc((size_t)DD * D_IN * 2);
  __bf16* Wt2_bf    = (__bf16*)alloc((size_t)DD * D_IN * 2);
  float*  feat1_f   = (float*) alloc((size_t)B_ROWS * DD * 4);
  float*  feat2_f   = (float*) alloc((size_t)B_ROWS * DD * 4);
  __bf16* feat1_bf  = (__bf16*)alloc((size_t)B_ROWS * DD * 2);
  __bf16* feat2_bf  = (__bf16*)alloc((size_t)B_ROWS * DD * 2);
  __bf16* feat1T_bf = (__bf16*)alloc((size_t)DD * B_ROWS * 2);
  __bf16* feat2T_bf = (__bf16*)alloc((size_t)DD * B_ROWS * 2);
  float*  featraw   = (float*) alloc((size_t)B_ROWS * DD * 4);
  float*  logits    = (float*) alloc((size_t)B_ROWS * KK * 4);
  __bf16* assignT   = (__bf16*)alloc((size_t)KK * B_ROWS * 2);
  float*  normv     = (float*) alloc((size_t)KK * 4);
  float*  aggraw    = (float*) alloc((size_t)KK * DD * 4);
  float*  agg1_f    = (float*) alloc((size_t)KK * DD * 4);
  __bf16* agg1_bf   = (__bf16*)alloc((size_t)KK * DD * 2);
  float*  agg2_f    = (float*) alloc((size_t)KK * DD * 4);
  __bf16* agg2_bf   = (__bf16*)alloc((size_t)KK * DD * 2);

  // stage normalized / transposed bf16 operands
  rownorm_bf16_kernel<<<KK, 256, 0, stream>>>(ctx1, ctx1_bf);
  rownorm_bf16_kernel<<<KK, 256, 0, stream>>>(ctx2, ctx2_bf);
  rownorm_bf16_kernel<<<LQ, 256, 0, stream>>>(qn, qn_bf);
  queue_mix_norm_kernel<<<QK_ROWS, 256, 0, stream>>>(qk, qnz, queue_bf);
  transpose_w_kernel<<<D_IN, 256, 0, stream>>>(W1, Wt1_bf);
  transpose_w_kernel<<<D_IN, 256, 0, stream>>>(W2, Wt2_bf);

  // features
  gemm_feat_kernel<<<256, 256, 0, stream>>>(x1, Wt1_bf, featraw);
  feat_norm_kernel<<<B_ROWS, 256, 0, stream>>>(featraw, feat1_f, feat1_bf, feat1T_bf, nullptr);
  gemm_feat_kernel<<<256, 256, 0, stream>>>(x2, Wt2_bf, featraw);
  feat_norm_kernel<<<B_ROWS, 256, 0, stream>>>(featraw, feat2_f, feat2_bf, feat2T_bf, out_aggn2);

  // view 1 cross-rep
  gemm_logits_kernel<<<1024, 256, 0, stream>>>(feat1_bf, ctx1_bf, logits);
  gumbel_assign_kernel<<<B_ROWS, 256, 0, stream>>>(logits, u1a, u1b, assignT, out_assign);
  normalizer_kernel<<<KK, 256, 0, stream>>>(assignT, normv);
  gemm_agg_kernel<<<64, 256, 0, stream>>>(assignT, feat1T_bf, aggraw);
  agg_norm_kernel<<<KK, 256, 0, stream>>>(aggraw, normv, agg1_f, agg1_bf, nullptr);

  // view 2 cross-rep
  gemm_logits_kernel<<<1024, 256, 0, stream>>>(feat2_bf, ctx2_bf, logits);
  gumbel_assign_kernel<<<B_ROWS, 256, 0, stream>>>(logits, u2a, u2b, assignT, nullptr);
  normalizer_kernel<<<KK, 256, 0, stream>>>(assignT, normv);
  gemm_agg_kernel<<<64, 256, 0, stream>>>(assignT, feat2T_bf, aggraw);
  agg_norm_kernel<<<KK, 256, 0, stream>>>(aggraw, normv, agg2_f, agg2_bf, out_aggk2);

  // losses
  zero_kernel<<<1, 32, 0, stream>>>(out_loss);
  loss_ce_kernel<<<B_ROWS / 16, 128, 0, stream>>>(feat1_f, feat1_bf, feat2_f, qn_bf,
                                                  LQ, 1.0f / (float)B_ROWS, out_loss);
  loss_ce_kernel<<<KK / 16, 128, 0, stream>>>(agg1_f, agg1_bf, agg2_f, queue_bf,
                                              QK_ROWS, 1.0f / (float)KK, out_loss);
}
